// MMDLoss_56478819942820
// MI455X (gfx1250) — matrix operand order repaired and checked by reading, not compiled, not run on hardware
//
#include <hip/hip_runtime.h>
#include <hip/hip_bf16.h>

typedef __attribute__((ext_vector_type(16))) __bf16 v16bf;
typedef __attribute__((ext_vector_type(8)))  __bf16 v8bf;
typedef __attribute__((ext_vector_type(8)))  float  v8f;

#define NS   4096
#define NTOT 8192
#define DDIM 256

#define SHUF16(a, b) __builtin_shufflevector(a, b, 0,1,2,3,4,5,6,7,8,9,10,11,12,13,14,15)

// A operand: per-lane elements 0..7 at +0, 8..15 at +16 (elements)
static __device__ __forceinline__ v16bf load_a16(const __bf16* p) {
    v8bf x = *(const v8bf*)p;
    v8bf y = *(const v8bf*)(p + 16);
    return SHUF16(x, y);
}
// B operand: per-lane elements 0..7 at +0, 8..15 at +16 (elements)
static __device__ __forceinline__ v16bf load_b16(const __bf16* p) {
    v8bf x = *(const v8bf*)p;
    v8bf y = *(const v8bf*)(p + 16);
    return SHUF16(x, y);
}

// ---------------------------------------------------------------------------
// Kernel 1: convert rows to split-bf16 (hi + lo), row squared norms, and
// per-block column partial sums (deterministic: no atomics).
// 1024 blocks x 256 threads; each wave (32 lanes) handles one row.
// ---------------------------------------------------------------------------
__global__ void __launch_bounds__(256)
mmd_prep(const float* __restrict__ src, const float* __restrict__ tgt,
         __bf16* __restrict__ hi, __bf16* __restrict__ lo,
         float* __restrict__ sq, float* __restrict__ colpart) {
    const int lane = threadIdx.x & 31;
    const int wave = threadIdx.x >> 5;
    const int row  = blockIdx.x * 8 + wave;

    const float* rp = (row < NS) ? (src + (size_t)row * DDIM)
                                 : (tgt + (size_t)(row - NS) * DDIM);

    __shared__ float cs[8 * DDIM];   // [wave][col] contributions, no conflicts

    float4 p0 = *(const float4*)(rp + lane * 8);
    float4 p1 = *(const float4*)(rp + lane * 8 + 4);
    float x[8] = {p0.x, p0.y, p0.z, p0.w, p1.x, p1.y, p1.z, p1.w};

    v8bf h8, l8;
    float ssq = 0.f;
#pragma unroll
    for (int e = 0; e < 8; ++e) {
        float v  = x[e];
        __bf16 hb = (__bf16)v;
        float  hv = (float)hb;
        __bf16 lb = (__bf16)(v - hv);
        h8[e] = hb;
        l8[e] = lb;
        ssq += v * v;
        cs[wave * DDIM + lane * 8 + e] = v;
    }
    *(v8bf*)(hi + (size_t)row * DDIM + lane * 8) = h8;
    *(v8bf*)(lo + (size_t)row * DDIM + lane * 8) = l8;

    // wave32 reduction of the row squared-norm (fixed tree order)
#pragma unroll
    for (int off = 16; off > 0; off >>= 1)
        ssq += __shfl_xor(ssq, off, 32);
    if (lane == 0) sq[row] = ssq;

    __syncthreads();
    // column partial sums for this block, fixed accumulation order
    const int t = threadIdx.x;
    float c = 0.f;
#pragma unroll
    for (int w = 0; w < 8; ++w) c += cs[w * DDIM + t];
    colpart[(size_t)blockIdx.x * DDIM + t] = c;
}

// ---------------------------------------------------------------------------
// Kernel 2: bandwidth.  sum(l2) = 2n*sum(sq) - 2*||colsum||^2  (exact identity)
// Stores nib[m] = -1/bw_m  for the 5 kernels.  Single block, deterministic.
// ---------------------------------------------------------------------------
__global__ void __launch_bounds__(256)
mmd_bw(const float* __restrict__ sq, const float* __restrict__ colpart,
       float* __restrict__ nib) {
    const int t = threadIdx.x;
    __shared__ float redA[256];
    __shared__ float redB[256];

    float s = 0.f;                       // column t full sum
    for (int b = 0; b < 1024; ++b) s += colpart[(size_t)b * DDIM + t];
    redA[t] = s * s;

    float q = 0.f;                       // strided slice of sum(sq)
    for (int i = 0; i < 32; ++i) q += sq[t + 256 * i];
    redB[t] = q;
    __syncthreads();

    for (int off = 128; off > 0; off >>= 1) {
        if (t < off) { redA[t] += redA[t + off]; redB[t] += redB[t + off]; }
        __syncthreads();
    }
    if (t == 0) {
        double gramsum = (double)redA[0];
        double sumsq   = (double)redB[0];
        double n  = (double)NTOT;
        double suml2 = 2.0 * n * sumsq - 2.0 * gramsum;
        double bw = suml2 / (n * n - n);
        bw = bw / 4.0;                   // / KERNEL_MUL^(KERNEL_NUM/2) = 2^2
        for (int m = 0; m < 5; ++m)
            nib[m] = (float)(-1.0 / (bw * (double)(1 << m)));
    }
}

// ---------------------------------------------------------------------------
// Kernel 3: main pass.  Block = 256 threads = 8 waves; block tile 64(M)x128(N);
// each wave computes a 16x64 strip = 4 WMMA accumulators, K-loop step 32.
// Gram via split-bf16: g ~= ah*bh + ah*bl + al*bh   (3x v_wmma per tile/step).
// All base pointers hoisted out of the k-loop: with full unroll every operand
// load is a global_load_b128 with an immediate offset (no VALU in hot loop).
// WMMAs are interleaved across the 4 N-tiles so dependent accumulations are
// separated by 3 independent WMMAs (no XDL stalls).
// ---------------------------------------------------------------------------
__global__ void __launch_bounds__(256)
mmd_main(const __bf16* __restrict__ hi, const __bf16* __restrict__ lo,
         const float* __restrict__ sq, const float* __restrict__ nib,
         float* __restrict__ blockpart) {
    const int lane  = threadIdx.x & 31;
    const int wave  = threadIdx.x >> 5;
    const int waveM = wave & 3;          // 4 waves stacked in M
    const int waveN = wave >> 2;         // 2 waves in N
    const int m0 = blockIdx.y * 64  + waveM * 16;
    const int n0 = blockIdx.x * 128 + waveN * 64;
    const int lm = lane & 15;
    const int lh = lane >> 4;

    const float nb0 = nib[0], nb1 = nib[1], nb2 = nib[2], nb3 = nib[3], nb4 = nib[4];

    v8f acc[4] = {v8f{}, v8f{}, v8f{}, v8f{}};

    // Hoisted base pointers (loop-invariant).
    // A: lane holds row (m0+lm); lanes 0-15 -> K {0..7,16..23}, lanes 16-31 -> +8
    const __bf16* aph = hi + (size_t)(m0 + lm) * DDIM + lh * 8;
    const __bf16* apl = lo + (size_t)(m0 + lm) * DDIM + lh * 8;
    // B: lane holds column (n+lm); lanes 0-15 -> K {0..7,16..23}, lanes 16-31 -> +8
    const __bf16* bph[4];
    const __bf16* bpl[4];
#pragma unroll
    for (int nt = 0; nt < 4; ++nt) {
        const size_t roff = (size_t)(n0 + nt * 16 + lm) * DDIM + lh * 8;
        bph[nt] = hi + roff;
        bpl[nt] = lo + roff;
    }

#pragma unroll
    for (int kk = 0; kk < DDIM; kk += 32) {
        v16bf ah = load_a16(aph + kk);
        v16bf al = load_a16(apl + kk);
        v16bf bh[4], bl[4];
#pragma unroll
        for (int nt = 0; nt < 4; ++nt) {
            bh[nt] = load_b16(bph[nt] + kk);
            bl[nt] = load_b16(bpl[nt] + kk);
        }
        // three sweeps: dependent WMMAs on the same acc are 4 apart
#pragma unroll
        for (int nt = 0; nt < 4; ++nt)
            acc[nt] = __builtin_amdgcn_wmma_f32_16x16x32_bf16(
                false, ah, false, bh[nt], (short)0, acc[nt], false, false);
#pragma unroll
        for (int nt = 0; nt < 4; ++nt)
            acc[nt] = __builtin_amdgcn_wmma_f32_16x16x32_bf16(
                false, ah, false, bl[nt], (short)0, acc[nt], false, false);
#pragma unroll
        for (int nt = 0; nt < 4; ++nt)
            acc[nt] = __builtin_amdgcn_wmma_f32_16x16x32_bf16(
                false, al, false, bh[nt], (short)0, acc[nt], false, false);
    }

    // Epilogue: C/D layout -> VGPR e: lanes 0-15 M=m0+e, lanes 16-31 M=m0+8+e.
    float sqj[4];
    bool  jx[4];
#pragma unroll
    for (int nt = 0; nt < 4; ++nt) {
        int j = n0 + nt * 16 + lm;
        sqj[nt] = sq[j];
        jx[nt]  = (j < NS);
    }

    float tsum = 0.f;
#pragma unroll
    for (int e = 0; e < 8; ++e) {
        const int i = m0 + e + lh * 8;
        const float sqi = sq[i];
        const bool  ix  = (i < NS);
#pragma unroll
        for (int nt = 0; nt < 4; ++nt) {
            float g  = acc[nt][e];
            float l2 = sqi + sqj[nt] - 2.0f * g;
            float kv = __expf(l2 * nb0) + __expf(l2 * nb1) + __expf(l2 * nb2)
                     + __expf(l2 * nb3) + __expf(l2 * nb4);
            tsum += (ix == jx[nt]) ? kv : -kv;
        }
    }

    __shared__ float red[256];
    red[threadIdx.x] = tsum;
    __syncthreads();
    for (int off = 128; off > 0; off >>= 1) {
        if ((int)threadIdx.x < off) red[threadIdx.x] += red[threadIdx.x + off];
        __syncthreads();
    }
    if (threadIdx.x == 0)
        blockpart[(size_t)blockIdx.y * gridDim.x + blockIdx.x] = red[0];
}

// ---------------------------------------------------------------------------
// Kernel 4: final deterministic reduction of 8192 block partials.
// loss = (S_XX + S_YY - S_XY - S_YX) / ns^2
// ---------------------------------------------------------------------------
__global__ void __launch_bounds__(256)
mmd_fin(const float* __restrict__ blockpart, float* __restrict__ out) {
    const int t = threadIdx.x;
    float s = 0.f;
    for (int i = 0; i < 32; ++i) s += blockpart[t + 256 * i];
    __shared__ float red[256];
    red[t] = s;
    __syncthreads();
    for (int off = 128; off > 0; off >>= 1) {
        if (t < off) red[t] += red[t + off];
        __syncthreads();
    }
    if (t == 0) out[0] = red[0] * (1.0f / ((float)NS * (float)NS));
}

// ---------------------------------------------------------------------------
extern "C" void kernel_launch(void* const* d_in, const int* in_sizes, int n_in,
                              void* d_out, int out_size, void* d_ws, size_t ws_size,
                              hipStream_t stream) {
    const float* src = (const float*)d_in[0];
    const float* tgt = (const float*)d_in[1];
    float* out = (float*)d_out;

    // workspace layout (~9.5 MB)
    char* w = (char*)d_ws;
    __bf16* hi = (__bf16*)w;                                // 8192*256 bf16
    __bf16* lo = hi + (size_t)NTOT * DDIM;                  // 8192*256 bf16
    float* sq        = (float*)(lo + (size_t)NTOT * DDIM);  // 8192 f32
    float* colpart   = sq + NTOT;                           // 1024*256 f32
    float* nib       = colpart + 1024 * DDIM;               // 8 f32 (5 used)
    float* blockpart = nib + 8;                             // 8192 f32

    mmd_prep<<<1024, 256, 0, stream>>>(src, tgt, hi, lo, sq, colpart);
    mmd_bw<<<1, 256, 0, stream>>>(sq, colpart, nib);
    mmd_main<<<dim3(64, 128), 256, 0, stream>>>(hi, lo, sq, nib, blockpart);
    mmd_fin<<<1, 256, 0, stream>>>(blockpart, out);
}
